// fusionVGG19_21852793602783
// MI455X (gfx1250) — matrix operand order repaired and checked by reading, not compiled, not run on hardware
//
#include <hip/hip_runtime.h>
#include <stdint.h>

// ---------- types ----------
typedef __attribute__((ext_vector_type(16))) __bf16        v16bf;
typedef __attribute__((ext_vector_type(8)))  float         v8f;
typedef __attribute__((ext_vector_type(4)))  unsigned int  u32x4;

__device__ __forceinline__ unsigned short f2bf(float f) {
  unsigned u = __builtin_bit_cast(unsigned, f);
  unsigned r = u + 0x7FFFu + ((u >> 16) & 1u);      // round-to-nearest-even
  return (unsigned short)(r >> 16);
}
__device__ __forceinline__ float bf2f(unsigned short h) {
  unsigned u = ((unsigned)h) << 16;
  return __builtin_bit_cast(float, u);
}

// ---------- Tensor Data Mover (async 2D tile -> LDS) ----------
#if defined(__has_builtin)
#if __has_builtin(__builtin_amdgcn_tensor_load_to_lds)
#define HAVE_TDM 1
#endif
#endif

#ifdef HAVE_TDM
typedef unsigned int tdm_u4 __attribute__((ext_vector_type(4)));
typedef int          tdm_i8 __attribute__((ext_vector_type(8)));
typedef int          tdm_i4 __attribute__((ext_vector_type(4)));

// Load a [rows x cols] bf16 tile (row stride = stride_elems) from gaddr into
// LDS at lds_addr, row-major [rows][cols]. OOB (x>=tdim0 / y>=tdim1) reads 0.
__device__ __forceinline__ void tdm_load_tile_2d(unsigned lds_addr,
                                                 const void* gaddr,
                                                 unsigned rows, unsigned cols,
                                                 unsigned tdim0, unsigned tdim1,
                                                 unsigned stride_elems) {
  unsigned long long ga = (unsigned long long)gaddr;
  tdm_u4 g0;
  g0.x = 1u;                                             // count=1 (valid D#)
  g0.y = lds_addr;                                       // LDS byte address
  g0.z = (unsigned)(ga & 0xFFFFFFFFu);                   // global_addr[31:0]
  g0.w = (unsigned)((ga >> 32) & 0x01FFFFFFu) | 0x80000000u; // [56:32] | type=2
  tdm_i8 g1;
  g1[0] = (int)(1u << 16);                               // data_size=2B
  g1[1] = (int)((tdim0 & 0xFFFFu) << 16);                // tensor_dim0 lo16
  g1[2] = (int)(((tdim0 >> 16) & 0xFFFFu) | ((tdim1 & 0xFFFFu) << 16));
  g1[3] = (int)(((tdim1 >> 16) & 0xFFFFu) | ((cols & 0xFFFFu) << 16)); // tile_dim0
  g1[4] = (int)(rows & 0xFFFFu);                         // tile_dim1 (tile_dim2=0)
  g1[5] = (int)stride_elems;                             // tensor_dim0_stride lo32
  g1[6] = 0;
  g1[7] = 0;
  tdm_i4 z4 = {0, 0, 0, 0};
#if __clang_major__ >= 23
  tdm_i8 z8 = {0, 0, 0, 0, 0, 0, 0, 0};
  __builtin_amdgcn_tensor_load_to_lds(g0, g1, z4, z4, z8, 0);
#else
  __builtin_amdgcn_tensor_load_to_lds(g0, g1, z4, z4, 0);
#endif
}
#endif // HAVE_TDM

// =====================================================================
// Implicit-im2col GEMM conv:  out[m,n] = sum_k W[m,k] * im2col(in)[k,n]
// 256 threads = 8 waves; block tile = 64 Cout x 128 pixels; K chunk = 32.
// Wave (wm 0..3, wn 0..1): A frag 16x32 reused for FOUR 16x16 outputs.
// A tiles double-buffered via async TDM (issue chunk c+1, wait tensorcnt<=1)
// overlapping the B im2col gather and the previous chunk's WMMAs.
// =====================================================================
template <int KSZ>
__launch_bounds__(256)
__global__ void conv_wmma_bf16(const unsigned short* __restrict__ in,
                               const unsigned short* __restrict__ wgt,
                               const float* __restrict__ bias,
                               float* __restrict__ out,
                               int Cin, int Cout, int H, int W, int dil) {
  __shared__ __attribute__((aligned(16))) unsigned short As[2][64 * 32]; // [m][k]
  __shared__ __attribute__((aligned(16))) unsigned short Bs[128 * 32];   // [n][k]
  const int HW   = H * W;
  const int Ktot = Cin * KSZ * KSZ;
  const int pad  = dil * (KSZ - 1) / 2;

  const int tid  = threadIdx.x;
  const int lane = tid & 31;
  const int wave = tid >> 5;
  const int wm   = wave & 3;   // 4 waves in M
  const int wn   = wave >> 2;  // 2 waves in N (each covers 64 pixels)
  const int coutBase = blockIdx.y * 64;
  const int pixBase  = blockIdx.x * 128;

  // B staging map: thread -> (n = tid>>1 in 0..127, 16 consecutive k)
  const int bn  = tid >> 1;
  const int bk0 = (tid & 1) * 16;
  const int pix = pixBase + bn;
  const bool pv = pix < HW;
  int py = 0, px = 0;
  if (pv) { py = pix / W; px = pix - py * W; }

  const int nch = (Ktot + 31) >> 5;

#ifdef HAVE_TDM
  if (wave == 0) {
    tdm_load_tile_2d((unsigned)(size_t)&As[0][0],
                     wgt + (size_t)coutBase * Ktot,
                     64u, 32u, (unsigned)Ktot, (unsigned)(Cout - coutBase),
                     (unsigned)Ktot);
  }
#endif

  v8f acc[4] = {};
  for (int c = 0; c < nch; ++c) {
    const int kk = c << 5;
#ifndef HAVE_TDM
    for (int i = tid; i < 64 * 32; i += 256) {
      int row = i >> 5, col = i & 31;
      int k = kk + col;
      As[0][i] = (k < Ktot) ? wgt[(size_t)(coutBase + row) * Ktot + k]
                            : (unsigned short)0;
    }
#endif
    // ---- B tile: im2col gather [n][k], 16 elems/thread ----
    {
      int k = kk + bk0;
      if (KSZ == 1) {
#pragma unroll
        for (int j = 0; j < 16; ++j) {
          unsigned short v = 0;
          if (k + j < Ktot && pv) v = in[(size_t)(k + j) * HW + pix];
          Bs[bn * 32 + bk0 + j] = v;
        }
      } else {
        // one division per thread per chunk, then incremental (kx,ky,ci)
        int ci = k / (KSZ * KSZ);
        int rem = k - ci * (KSZ * KSZ);
        int ky = rem / KSZ;
        int kx = rem - ky * KSZ;
#pragma unroll
        for (int j = 0; j < 16; ++j) {
          unsigned short v = 0;
          if (k + j < Ktot && pv) {
            int iy = py + ky * dil - pad;
            int ix = px + kx * dil - pad;
            if (iy >= 0 && iy < H && ix >= 0 && ix < W)
              v = in[(size_t)ci * HW + iy * W + ix];
          }
          Bs[bn * 32 + bk0 + j] = v;
          if (++kx == KSZ) { kx = 0; if (++ky == KSZ) { ky = 0; ++ci; } }
        }
      }
    }
#ifdef HAVE_TDM
    if (wave == 0) {
      if (c + 1 < nch) {
        tdm_load_tile_2d((unsigned)(size_t)&As[(c + 1) & 1][0],
                         wgt + (size_t)coutBase * Ktot + (kk + 32),
                         64u, 32u, (unsigned)(Ktot - (kk + 32)),
                         (unsigned)(Cout - coutBase), (unsigned)Ktot);
        __builtin_amdgcn_s_wait_tensorcnt(1);  // current tile landed
      } else {
        __builtin_amdgcn_s_wait_tensorcnt(0);
      }
    }
#endif
    __syncthreads();

#ifdef HAVE_TDM
    const unsigned short* Acur = &As[c & 1][0];
#else
    const unsigned short* Acur = &As[0][0];
#endif
    // ---- fragments (ISA 16-bit A 16x32 / B 32x16 VGPR layouts) ----
    const int g = lane >> 4, r = lane & 15;
    const u32x4* arow = (const u32x4*)&Acur[(wm * 16 + r) * 32];
    union { u32x4 u[2]; v16bf v; } fa;
    fa.u[0] = arow[g];
    fa.u[1] = arow[2 + g];
#pragma unroll
    for (int sub = 0; sub < 4; ++sub) {
      const u32x4* br = (const u32x4*)&Bs[(wn * 64 + sub * 16 + r) * 32];
      union { u32x4 u[2]; v16bf v; } fb;
      fb.u[0] = br[g];
      fb.u[1] = br[2 + g];
      acc[sub] = __builtin_amdgcn_wmma_f32_16x16x32_bf16(
          false, fa.v, false, fb.v, (short)0, acc[sub], false, false);
    }
    __syncthreads();
  }

  // ---- store C (lanes 0-15: M=e, lanes 16-31: M=e+8) ----
  const int g = lane >> 4, r = lane & 15;
#pragma unroll
  for (int e = 0; e < 8; ++e) {
    int m = coutBase + wm * 16 + g * 8 + e;
    float bb = bias ? bias[m] : 0.f;
#pragma unroll
    for (int sub = 0; sub < 4; ++sub) {
      int n = pixBase + wn * 64 + sub * 16 + r;
      if (n < HW) out[(size_t)m * HW + n] = acc[sub][e] + bb;
    }
  }
}

// ---------------- elementwise / reduction helpers ----------------
__global__ void cast_f32_bf16_kernel(const float* __restrict__ in,
                                     unsigned short* __restrict__ out, int n) {
  int i = blockIdx.x * blockDim.x + threadIdx.x;
  if (i < n) out[i] = f2bf(in[i]);
}

__global__ void bn_stats_kernel(const float* __restrict__ x,
                                float* __restrict__ mean,
                                float* __restrict__ rstd, int HW) {
  __shared__ float s1[256], s2[256];
  const int c = blockIdx.x;
  const float* p = x + (size_t)c * HW;
  float a = 0.f, b = 0.f;
  for (int i = threadIdx.x; i < HW; i += 256) { float v = p[i]; a += v; b += v * v; }
  s1[threadIdx.x] = a; s2[threadIdx.x] = b; __syncthreads();
  for (int s = 128; s > 0; s >>= 1) {
    if (threadIdx.x < s) { s1[threadIdx.x] += s1[threadIdx.x + s];
                           s2[threadIdx.x] += s2[threadIdx.x + s]; }
    __syncthreads();
  }
  if (threadIdx.x == 0) {
    float m = s1[0] / HW;
    float v = s2[0] / HW - m * m;
    mean[c] = m;
    rstd[c] = rsqrtf(v + 1e-5f);
  }
}

__global__ void bn_relu_bf16_kernel(const float* __restrict__ x,
                                    unsigned short* __restrict__ y,
                                    const float* __restrict__ mean,
                                    const float* __restrict__ rstd,
                                    const float* __restrict__ g,
                                    const float* __restrict__ b,
                                    int HW, int n) {
  int i = blockIdx.x * blockDim.x + threadIdx.x;
  if (i >= n) return;
  int c = i / HW;
  float v = (x[i] - mean[c]) * rstd[c] * g[c] + b[c];
  y[i] = f2bf(v > 0.f ? v : 0.f);
}

__global__ void maxpool2_kernel(const unsigned short* __restrict__ in,
                                unsigned short* __restrict__ out,
                                int C, int H, int W) {
  int Ho = H >> 1, Wo = W >> 1;
  int n = C * Ho * Wo;
  int i = blockIdx.x * blockDim.x + threadIdx.x;
  if (i >= n) return;
  int c = i / (Ho * Wo), rem = i % (Ho * Wo);
  int y = rem / Wo, x = rem % Wo;
  const unsigned short* p = in + (size_t)c * H * W + (2 * y) * W + 2 * x;
  float a = bf2f(p[0]), b = bf2f(p[1]), cc = bf2f(p[W]), d = bf2f(p[W + 1]);
  out[i] = f2bf(fmaxf(fmaxf(a, b), fmaxf(cc, d)));
}

__device__ __forceinline__ float bilin_bf16(const unsigned short* p, int h, int w,
                                            float fy, float fx) {
  int y0 = (int)floorf(fy), x0 = (int)floorf(fx);
  float wy = fy - (float)y0, wx = fx - (float)x0;
  int y1 = y0 + 1, x1 = x0 + 1;
  y0 = min(max(y0, 0), h - 1); y1 = min(max(y1, 0), h - 1);
  x0 = min(max(x0, 0), w - 1); x1 = min(max(x1, 0), w - 1);
  float a = bf2f(p[y0 * w + x0]), b = bf2f(p[y0 * w + x1]);
  float c = bf2f(p[y1 * w + x0]), d = bf2f(p[y1 * w + x1]);
  return (1.f - wy) * ((1.f - wx) * a + wx * b) + wy * ((1.f - wx) * c + wx * d);
}

__global__ void upsample_bf16_kernel(const unsigned short* __restrict__ in,
                                     unsigned short* __restrict__ out,
                                     int C, int h, int w, int s) {
  int Ho = h * s, Wo = w * s;
  int n = C * Ho * Wo;
  int i = blockIdx.x * blockDim.x + threadIdx.x;
  if (i >= n) return;
  int c = i / (Ho * Wo), rem = i % (Ho * Wo);
  int y = rem / Wo, x = rem % Wo;
  float fy = (y + 0.5f) / s - 0.5f;
  float fx = (x + 0.5f) / s - 0.5f;
  out[i] = f2bf(bilin_bf16(in + (size_t)c * h * w, h, w, fy, fx));
}

__global__ void avgpool8_bf16_kernel(const unsigned short* __restrict__ in,
                                     unsigned short* __restrict__ out,
                                     int C, int H, int W) {
  int Ho = H >> 3, Wo = W >> 3;
  int n = C * Ho * Wo;
  int i = blockIdx.x * blockDim.x + threadIdx.x;
  if (i >= n) return;
  int c = i / (Ho * Wo), rem = i % (Ho * Wo);
  int y = rem / Wo, x = rem % Wo;
  const unsigned short* p = in + (size_t)c * H * W + (8 * y) * W + 8 * x;
  float s = 0.f;
  for (int dy = 0; dy < 8; ++dy)
    for (int dx = 0; dx < 8; ++dx) s += bf2f(p[dy * W + dx]);
  out[i] = f2bf(s * (1.f / 64.f));
}

// transpose + cast with column zero-pad: out[k*Npad+n] = in[n*K+k] (n<N)
__global__ void transpose_cast_kernel(const float* __restrict__ in,
                                      unsigned short* __restrict__ out,
                                      int N, int K, int Npad) {
  int total = K * Npad;
  int i = blockIdx.x * blockDim.x + threadIdx.x;
  if (i >= total) return;
  int k = i / Npad, n = i % Npad;
  out[i] = (n < N) ? f2bf(in[(size_t)n * K + k]) : (unsigned short)0;
}

__global__ void reduce_all_kernel(const float* __restrict__ x, int n,
                                  float* __restrict__ out2) {
  __shared__ float s1[256], s2[256];
  float a = 0.f, b = 0.f;
  for (int i = threadIdx.x; i < n; i += 256) { float v = x[i]; a += v; b += v * v; }
  s1[threadIdx.x] = a; s2[threadIdx.x] = b; __syncthreads();
  for (int s = 128; s > 0; s >>= 1) {
    if (threadIdx.x < s) { s1[threadIdx.x] += s1[threadIdx.x + s];
                           s2[threadIdx.x] += s2[threadIdx.x + s]; }
    __syncthreads();
  }
  if (threadIdx.x == 0) {
    float m = s1[0] / n;
    float v = s2[0] / n - m * m;
    out2[0] = m;
    out2[1] = rsqrtf(v + 1e-5f);
  }
}

__global__ void tanh_bn_kernel(const float* __restrict__ x,
                               unsigned short* __restrict__ y,
                               const float* __restrict__ st,
                               const float* __restrict__ g,
                               const float* __restrict__ b, int n) {
  int i = blockIdx.x * blockDim.x + threadIdx.x;
  if (i >= n) return;
  float v = (x[i] - st[0]) * st[1] * g[0] + b[0];
  y[i] = f2bf(tanhf(v));
}

__global__ void zero_u16_kernel(unsigned short* p, int n) {
  int i = blockIdx.x * blockDim.x + threadIdx.x;
  if (i < n) p[i] = 0;
}

// softmax over M=256 for column j, folded into effective pred weights
__global__ void softmax_weff_kernel(const float* __restrict__ L,     // [256,64]
                                    const float* __restrict__ predw, // [57,256]
                                    unsigned short* __restrict__ weff) { // [64,256]
  __shared__ float red[256];
  const int j = blockIdx.x;      // 0..56
  const int m = threadIdx.x;     // 0..255
  float v = L[m * 64 + j];
  red[m] = v; __syncthreads();
  for (int s = 128; s > 0; s >>= 1) {
    if (m < s) red[m] = fmaxf(red[m], red[m + s]);
    __syncthreads();
  }
  float mx = red[0]; __syncthreads();
  float e = expf(v - mx);
  red[m] = e; __syncthreads();
  for (int s = 128; s > 0; s >>= 1) {
    if (m < s) red[m] += red[m + s];
    __syncthreads();
  }
  float sum = red[0];
  weff[j * 256 + m] = f2bf(predw[j * 256 + m] * (e / sum) * 256.0f);
}

__global__ void upsample_out_kernel(const float* __restrict__ pred, // [>=57,32000]
                                    const float* __restrict__ bias,
                                    float* __restrict__ out) {      // [57,800,640]
  const int HO = 800, WO = 640, hi = 200, wi = 160;
  int n = 57 * HO * WO;
  int i = blockIdx.x * blockDim.x + threadIdx.x;
  if (i >= n) return;
  int j = i / (HO * WO), rem = i % (HO * WO);
  int y = rem / WO, x = rem % WO;
  float fy = (y + 0.5f) * 0.25f - 0.5f;
  float fx = (x + 0.5f) * 0.25f - 0.5f;
  int y0 = (int)floorf(fy), x0 = (int)floorf(fx);
  float wy = fy - (float)y0, wx = fx - (float)x0;
  int y1 = y0 + 1, x1 = x0 + 1;
  y0 = min(max(y0, 0), hi - 1); y1 = min(max(y1, 0), hi - 1);
  x0 = min(max(x0, 0), wi - 1); x1 = min(max(x1, 0), wi - 1);
  const float* p = pred + (size_t)j * hi * wi;
  float a = p[y0 * wi + x0], b = p[y0 * wi + x1];
  float c = p[y1 * wi + x0], d = p[y1 * wi + x1];
  float v = (1.f - wy) * ((1.f - wx) * a + wx * b) + wy * ((1.f - wx) * c + wx * d);
  out[i] = v + bias[j];
}

// =====================================================================
// Host orchestration
// =====================================================================
extern "C" void kernel_launch(void* const* d_in, const int* in_sizes, int n_in,
                              void* d_out, int out_size, void* d_ws, size_t ws_size,
                              hipStream_t stream) {
  (void)in_sizes; (void)n_in; (void)out_size; (void)ws_size;

  static const int VIN[16]  = {3,64,64,128,128,256,256,256,256,512,512,512,512,512,512,512};
  static const int VOUT[16] = {64,64,128,128,256,256,256,256,512,512,512,512,512,512,512,512};
  static const int FCIN[4]  = {128,256,512,512};
  static const int TAPH[4]  = {200,100,50,25};
  static const int TAPW[4]  = {160,80,40,20};
  static const int DKS[4]   = {1,3,3,3};
  static const int DDIL[4]  = {1,1,2,4};

  const float* x = (const float*)d_in[0];
  const float *vw[16], *vb[16], *vg[16], *vbe[16];
  for (int i = 0; i < 16; ++i) {
    vw[i]  = (const float*)d_in[1 + i];
    vb[i]  = (const float*)d_in[17 + i];
    vg[i]  = (const float*)d_in[33 + i];
    vbe[i] = (const float*)d_in[49 + i];
  }
  const float *fw[4], *fb[4], *fg[4], *fbe[4];
  for (int j = 0; j < 4; ++j) {
    fw[j]  = (const float*)d_in[65 + j];
    fb[j]  = (const float*)d_in[69 + j];
    fg[j]  = (const float*)d_in[73 + j];
    fbe[j] = (const float*)d_in[77 + j];
  }
  const float *dw[4], *db[4], *dg[4], *dbe[4];
  for (int j = 0; j < 4; ++j) {
    dw[j]  = (const float*)d_in[81 + j];
    db[j]  = (const float*)d_in[85 + j];
    dg[j]  = (const float*)d_in[89 + j];
    dbe[j] = (const float*)d_in[93 + j];
  }
  const float* att_w1 = (const float*)d_in[97];
  const float* att_g  = (const float*)d_in[98];
  const float* att_b  = (const float*)d_in[99];
  const float* att_w2 = (const float*)d_in[100];
  const float* pred_w = (const float*)d_in[101];
  const float* pred_b = (const float*)d_in[102];

  // ---- bump allocator over d_ws ----
  char* base = (char*)d_ws;
  size_t off = 0;
  auto alloc = [&](size_t bytes) -> char* {
    char* p = base + off;
    off += (bytes + 255) & ~(size_t)255;
    return p;
  };

  const size_t MAXACT = (size_t)64 * 800 * 640; // 32.768M elems
  unsigned short* actA = (unsigned short*)alloc(MAXACT * 2);
  unsigned short* actB = (unsigned short*)alloc(MAXACT * 2);
  float*          convf = (float*)alloc(MAXACT * 4);
  float* mean = (float*)alloc(512 * 4);
  float* rstd = (float*)alloc(512 * 4);

  // weight casts to bf16
  unsigned short* vwb[16];
  for (int i = 0; i < 16; ++i) {
    int cnt = VOUT[i] * VIN[i] * 9;
    vwb[i] = (unsigned short*)alloc((size_t)cnt * 2);
    cast_f32_bf16_kernel<<<(cnt + 255) / 256, 256, 0, stream>>>(vw[i], vwb[i], cnt);
  }
  unsigned short* fwb[4];
  for (int j = 0; j < 4; ++j) {
    int cnt = 64 * FCIN[j];
    fwb[j] = (unsigned short*)alloc((size_t)cnt * 2);
    cast_f32_bf16_kernel<<<(cnt + 255) / 256, 256, 0, stream>>>(fw[j], fwb[j], cnt);
  }
  unsigned short* dwb[4];
  for (int j = 0; j < 4; ++j) {
    int cnt = 64 * 256 * DKS[j] * DKS[j];
    dwb[j] = (unsigned short*)alloc((size_t)cnt * 2);
    cast_f32_bf16_kernel<<<(cnt + 255) / 256, 256, 0, stream>>>(dw[j], dwb[j], cnt);
  }
  unsigned short* taps[4];
  for (int j = 0; j < 4; ++j)
    taps[j] = (unsigned short*)alloc((size_t)FCIN[j] * TAPH[j] * TAPW[j] * 2);

  // ---- backbone ----
  int Hc = 800, Wc = 640;
  {
    int n = 3 * 800 * 640;
    cast_f32_bf16_kernel<<<(n + 255) / 256, 256, 0, stream>>>(x, actB, n);
  }
  const unsigned short* cur = actB;
  int tap = 0;
  for (int i = 0; i < 16; ++i) {
    const int HW = Hc * Wc, Cout = VOUT[i];
    dim3 grid((HW + 127) / 128, Cout / 64);
    conv_wmma_bf16<3><<<grid, 256, 0, stream>>>(cur, vwb[i], vb[i], convf,
                                                VIN[i], Cout, Hc, Wc, 1);
    bn_stats_kernel<<<Cout, 256, 0, stream>>>(convf, mean, rstd, HW);
    int n = Cout * HW;
    bn_relu_bf16_kernel<<<(n + 255) / 256, 256, 0, stream>>>(
        convf, actA, mean, rstd, vg[i], vbe[i], HW, n);
    cur = actA;
    const bool pool = (i == 1 || i == 3 || i == 7 || i == 11 || i == 15);
    const bool tapped = (i == 3 || i == 7 || i == 11 || i == 15);
    if (pool) {
      unsigned short* pd = tapped ? taps[tap] : actB;
      int n2 = Cout * (Hc / 2) * (Wc / 2);
      maxpool2_kernel<<<(n2 + 255) / 256, 256, 0, stream>>>(actA, pd, Cout, Hc, Wc);
      Hc >>= 1; Wc >>= 1;
      cur = pd;
      if (tapped) ++tap;
    }
  }

  // ---- feature convs (1x1) + BN + ReLU ----
  unsigned short* fsb[4];
  for (int j = 0; j < 4; ++j) {
    const int HW = TAPH[j] * TAPW[j];
    fsb[j] = (unsigned short*)alloc((size_t)64 * HW * 2);
    dim3 grid((HW + 127) / 128, 1);
    conv_wmma_bf16<1><<<grid, 256, 0, stream>>>(taps[j], fwb[j], fb[j], convf,
                                                FCIN[j], 64, TAPH[j], TAPW[j], 1);
    bn_stats_kernel<<<64, 256, 0, stream>>>(convf, mean, rstd, HW);
    int n = 64 * HW;
    bn_relu_bf16_kernel<<<(n + 255) / 256, 256, 0, stream>>>(
        convf, fsb[j], mean, rstd, fg[j], fbe[j], HW, n);
  }

  // ---- bone assembly: upsample to 200x160, concat channels ----
  const int BHW = 200 * 160; // 32000
  unsigned short* bone  = (unsigned short*)alloc((size_t)256 * BHW * 2);
  unsigned short* bone2 = (unsigned short*)alloc((size_t)256 * BHW * 2);
  for (int j = 0; j < 4; ++j) {
    int s = 1 << j;
    int n = 64 * BHW;
    upsample_bf16_kernel<<<(n + 255) / 256, 256, 0, stream>>>(
        fsb[j], bone + (size_t)j * 64 * BHW, 64, TAPH[j], TAPW[j], s);
  }

  // ---- dilated inception block ----
  for (int j = 0; j < 4; ++j) {
    dim3 grid((BHW + 127) / 128, 1);
    if (DKS[j] == 1)
      conv_wmma_bf16<1><<<grid, 256, 0, stream>>>(bone, dwb[j], db[j], convf,
                                                  256, 64, 200, 160, 1);
    else
      conv_wmma_bf16<3><<<grid, 256, 0, stream>>>(bone, dwb[j], db[j], convf,
                                                  256, 64, 200, 160, DDIL[j]);
    bn_stats_kernel<<<64, 256, 0, stream>>>(convf, mean, rstd, BHW);
    int n = 64 * BHW;
    bn_relu_bf16_kernel<<<(n + 255) / 256, 256, 0, stream>>>(
        convf, bone2 + (size_t)j * 64 * BHW, mean, rstd, dg[j], dbe[j], BHW, n);
  }

  // ---- attention ----
  unsigned short* apool = (unsigned short*)alloc((size_t)256 * 500 * 2);
  {
    int n = 256 * 500;
    avgpool8_bf16_kernel<<<(n + 255) / 256, 256, 0, stream>>>(bone2, apool, 256, 200, 160);
  }
  unsigned short* w1t = (unsigned short*)alloc((size_t)500 * 128 * 2);
  {
    int n = 500 * 128;
    transpose_cast_kernel<<<(n + 255) / 256, 256, 0, stream>>>(att_w1, w1t, 128, 500, 128);
  }
  float* hhf = (float*)alloc((size_t)256 * 128 * 4);
  {
    // hh[256,128] = a[256,500] @ w1^T  (weight=a, input=w1t [500,128])
    dim3 grid(1, 256 / 64);
    conv_wmma_bf16<1><<<grid, 256, 0, stream>>>(w1t, apool, nullptr, hhf,
                                                500, 256, 1, 128, 1);
  }
  float* gstat = (float*)alloc(8);
  reduce_all_kernel<<<1, 256, 0, stream>>>(hhf, 256 * 128, gstat);
  unsigned short* hhbf = (unsigned short*)alloc((size_t)256 * 128 * 2);
  tanh_bn_kernel<<<(256 * 128 + 255) / 256, 256, 0, stream>>>(
      hhf, hhbf, gstat, att_g, att_b, 256 * 128);

  unsigned short* w2t = (unsigned short*)alloc((size_t)128 * 64 * 2);
  {
    int n = 128 * 64;
    transpose_cast_kernel<<<(n + 255) / 256, 256, 0, stream>>>(att_w2, w2t, 57, 128, 64);
  }
  float* Lf = (float*)alloc((size_t)256 * 64 * 4);
  {
    // L[256,64] = hh[256,128] @ w2^T (padded to 64 cols)
    dim3 grid(1, 256 / 64);
    conv_wmma_bf16<1><<<grid, 256, 0, stream>>>(w2t, hhbf, nullptr, Lf,
                                                128, 256, 1, 64, 1);
  }
  unsigned short* weff = (unsigned short*)alloc((size_t)64 * 256 * 2);
  zero_u16_kernel<<<(64 * 256 + 255) / 256, 256, 0, stream>>>(weff, 64 * 256);
  softmax_weff_kernel<<<57, 256, 0, stream>>>(Lf, pred_w, weff);

  // ---- prediction (57x256 GEMM over 32000 pixels, rows padded to 64) ----
  float* predf = (float*)alloc((size_t)64 * BHW * 4);
  {
    dim3 grid((BHW + 127) / 128, 1);
    conv_wmma_bf16<1><<<grid, 256, 0, stream>>>(bone2, weff, nullptr, predf,
                                                256, 64, 200, 160, 1);
  }

  // ---- final x4 bilinear upsample + bias into d_out ----
  {
    int n = 57 * 800 * 640;
    upsample_out_kernel<<<(n + 255) / 256, 256, 0, stream>>>(
        predf, pred_b, (float*)d_out);
  }
}
